// selfattention_40553081209247
// MI455X (gfx1250) — matrix operand order repaired and checked by reading, not compiled, not run on hardware
//
#include <hip/hip_runtime.h>

// ---------------------------------------------------------------------------
// Multi-head self-attention (B=4,S=2048,E=1024,H=16,D=64), unmasked softmax.
// Compute-bound (~138 GFLOP vs ~150MB HBM) -> all GEMMs on
// v_wmma_f32_16x16x32_bf16. q/k/v (48MB bf16) are L2-resident (192MB L2).
// GEMMs use 32x64 register blocking per wave: 8 WMMA per 6 fragment loads.
// ---------------------------------------------------------------------------

typedef __bf16 bf16;
typedef __attribute__((ext_vector_type(16))) __bf16 v16bf;
typedef __attribute__((ext_vector_type(8)))  __bf16 v8bf;
typedef __attribute__((ext_vector_type(4)))  __bf16 v4bf;
typedef __attribute__((ext_vector_type(8)))  float  v8f;
typedef __attribute__((ext_vector_type(4)))  float  v4f;

#define B_  4
#define S_  2048
#define E_  1024
#define H_  16
#define D_  64
#define N3_ 3072

__device__ __forceinline__ bf16 f2bf(float f) {
  unsigned u = __builtin_bit_cast(unsigned, f);
  unsigned r = (u + 0x7fffu + ((u >> 16) & 1u)) >> 16;   // round-to-nearest-even
  return __builtin_bit_cast(bf16, (unsigned short)r);
}
__device__ __forceinline__ v8bf ld8(const bf16* p) { return *(const v8bf*)p; }
__device__ __forceinline__ v16bf cat8(v8bf a, v8bf b) {
  return __builtin_shufflevector(a, b, 0,1,2,3,4,5,6,7,8,9,10,11,12,13,14,15);
}
// A-fragment (16x32 bf16): lane row fixed; ks = {0..15} (lanes 0-15)
// or {16..31} (lanes 16-31). Two 16B loads.
__device__ __forceinline__ v16bf frag_a(const bf16* row, int half) {
  return cat8(ld8(row + half * 16), ld8(row + half * 16 + 8));
}
// B-fragment (32x16 bf16): lane col = lane%16; K = half*16 .. +16 contiguous.
__device__ __forceinline__ v16bf frag_b(const bf16* p) {
  return cat8(ld8(p), ld8(p + 8));
}
__device__ __forceinline__ v8f wmma_bf16(v16bf a, v16bf b, v8f c) {
  return __builtin_amdgcn_wmma_f32_16x16x32_bf16(false, a, false, b,
                                                 (short)0, c, false, false);
}

// ---- pass 0a: fp32 -> bf16 (same layout), 4 elems/thread --------------------
__global__ void cvt4_kernel(const float* __restrict__ in, bf16* __restrict__ out,
                            int n4) {
  int i = blockIdx.x * blockDim.x + threadIdx.x;
  if (i < n4) {
    v4f f = *(const v4f*)(in + (size_t)i * 4);
    v4bf o;
    o[0] = f2bf(f[0]); o[1] = f2bf(f[1]); o[2] = f2bf(f[2]); o[3] = f2bf(f[3]);
    *(v4bf*)(out + (size_t)i * 4) = o;
  }
}

// ---- pass 0b: fp32 [R][C] -> bf16 [C][R] (transpose+convert) ----------------
__global__ void tcvt_kernel(const float* __restrict__ in, bf16* __restrict__ out,
                            int R, int C) {
  int i = blockIdx.x * blockDim.x + threadIdx.x;
  if (i < R * C) {
    int r = i / C, c = i % C;
    out[(size_t)c * R + r] = f2bf(in[i]);
  }
}

// ---- pass 1: QKV GEMM  [8192x1024] x [1024x3072] + b_in ---------------------
// 32(M) x 64(N) per wave: 2 A-frags x 4 B-frags -> 8 WMMA per k-step.
// Writes q,k as [B,H,S,D] bf16 and v transposed as [B,H,D,S] bf16.
__global__ void __launch_bounds__(256)
qkv_gemm_kernel(const bf16* __restrict__ xb, const bf16* __restrict__ wt,
                const float* __restrict__ bias, bf16* __restrict__ q,
                bf16* __restrict__ kbuf, bf16* __restrict__ vT) {
  const int NT = N3_ / 64;                       // 48 col blocks
  int lane = threadIdx.x & 31, wave = threadIdx.x >> 5;
  int half = lane >> 4, l16 = lane & 15;
  int tile = blockIdx.x * 8 + wave;
  int mT = tile / NT, nT = tile % NT;
  int m0 = mT * 32, n0 = nT * 64;
  const bf16* arow0 = xb + (size_t)(m0 + l16) * E_;
  const bf16* arow1 = xb + (size_t)(m0 + 16 + l16) * E_;
  const bf16* bptr  = wt + (size_t)(n0 + l16) * E_;   // wt is [N][K]
  v8f acc[2][4] = {};
#pragma unroll 2
  for (int k0 = 0; k0 < E_; k0 += 32) {
    v16bf a0 = frag_a(arow0 + k0, half);
    v16bf a1 = frag_a(arow1 + k0, half);
    const bf16* bp = bptr + k0 + half * 16;
    v16bf b0 = frag_b(bp);
    v16bf b1 = frag_b(bp + 16 * E_);
    v16bf b2 = frag_b(bp + 32 * E_);
    v16bf b3 = frag_b(bp + 48 * E_);
    acc[0][0] = wmma_bf16(a0, b0, acc[0][0]);
    acc[0][1] = wmma_bf16(a0, b1, acc[0][1]);
    acc[0][2] = wmma_bf16(a0, b2, acc[0][2]);
    acc[0][3] = wmma_bf16(a0, b3, acc[0][3]);
    acc[1][0] = wmma_bf16(a1, b0, acc[1][0]);
    acc[1][1] = wmma_bf16(a1, b1, acc[1][1]);
    acc[1][2] = wmma_bf16(a1, b2, acc[1][2]);
    acc[1][3] = wmma_bf16(a1, b3, acc[1][3]);
  }
#pragma unroll
  for (int mi = 0; mi < 2; mi++) {
#pragma unroll
    for (int j = 0; j < 4; j++) {
      int col = n0 + j * 16 + l16;
      float bv = bias[col];
      int sect = col >> 10;                      // 0=q 1=k 2=v (uniform/16 lanes)
      int e = col & (E_ - 1);
      int h = e >> 6, d = e & 63;
#pragma unroll
      for (int i = 0; i < 8; i++) {
        int row = m0 + mi * 16 + i + half * 8;   // C layout: M = vgpr + half*8
        int bb = row >> 11, s = row & (S_ - 1);
        bf16 o = f2bf(acc[mi][j][i] + bv);
        size_t hid = (size_t)bb * H_ + h;
        if (sect == 0)      q   [(hid * S_ + s) * D_ + d] = o;
        else if (sect == 1) kbuf[(hid * S_ + s) * D_ + d] = o;
        else                vT  [(hid * D_ + d) * S_ + s] = o;
      }
    }
  }
}

// ---- pass 2: flash attention (online softmax), 8 waves x 16 q-rows ----------
// 64 keys per iteration: 8 score WMMAs + 8 attn*V WMMAs per softmax pass.
__global__ void __launch_bounds__(256)
attn_kernel(const bf16* __restrict__ q, const bf16* __restrict__ k,
            const bf16* __restrict__ vT, bf16* __restrict__ ao) {
  __shared__ __align__(16) float sc[8][16][64];      // score staging, per wave
  __shared__ float salpha[8][16];
  __shared__ __align__(16) bf16 pbuf[8][16][64];     // exp weights (bf16)
  __shared__ float slinv[8][16];

  int lane = threadIdx.x & 31, wave = threadIdx.x >> 5;
  int half = lane >> 4, l16 = lane & 15;
  const int nqb = S_ / 128;
  int qb = blockIdx.x % nqb;
  int h  = (blockIdx.x / nqb) % H_;
  int b  = blockIdx.x / (nqb * H_);
  int q0 = qb * 128 + wave * 16;

  size_t hid = (size_t)b * H_ + h;
  const bf16* qrow = q  + (hid * S_ + q0 + l16) * D_;
  const bf16* kb   = k  + hid * S_ * D_;
  const bf16* vtb  = vT + hid * D_ * S_ + (size_t)l16 * S_;

  v16bf aq0 = frag_a(qrow, half);                // d = 0..31
  v16bf aq1 = frag_a(qrow + 32, half);           // d = 32..63
  v8f o0 = {}, o1 = {}, o2 = {}, o3 = {};
  float m_r = -3.0e38f, l_r = 0.f;               // valid on lanes 0..15
  const float scale = 0.125f;                    // 1/sqrt(64)

  for (int kt = 0; kt < S_; kt += 64) {
    // scores: Q(16x64) x K^T -> four 16x16 key tiles, K-dim (=d) split 2x32
#pragma unroll
    for (int t = 0; t < 4; t++) {
      const bf16* kr = kb + (size_t)(kt + t * 16 + l16) * D_;
      v8f s = {};
      s = wmma_bf16(aq0, frag_b(kr + half * 16), s);
      s = wmma_bf16(aq1, frag_b(kr + 32 + half * 16), s);
#pragma unroll
      for (int i = 0; i < 8; i++)
        sc[wave][i + half * 8][t * 16 + l16] = s[i] * scale;
    }
    __syncthreads();
    if (lane < 16) {                             // one lane per q-row
      const float* srow = sc[wave][lane];
      float tmax = srow[0];
      for (int j = 1; j < 64; j++) tmax = fmaxf(tmax, srow[j]);
      float m_new = fmaxf(m_r, tmax);
      float alpha = __expf(m_r - m_new);
      float sum = 0.f;
      for (int j = 0; j < 64; j++) {
        float p = __expf(srow[j] - m_new);
        pbuf[wave][lane][j] = f2bf(p);
        sum += p;
      }
      l_r = l_r * alpha + sum;
      m_r = m_new;
      salpha[wave][lane] = alpha;
    }
    __syncthreads();
#pragma unroll
    for (int i = 0; i < 8; i++) {
      float a = salpha[wave][i + half * 8];
      o0[i] *= a; o1[i] *= a; o2[i] *= a; o3[i] *= a;
    }
    // O += P(16x64) x V(64x64): two A-frags from LDS, eight B-frags from vT
    v16bf pa0 = frag_a(&pbuf[wave][l16][0], half);        // keys kt+0..31
    v16bf pa1 = frag_a(&pbuf[wave][l16][32], half);       // keys kt+32..63
    const bf16* vr0 = vtb + kt + half * 16;               // keys kt+0..31
    const bf16* vr1 = vtb + kt + 32 + half * 16;          // keys kt+32..63
    o0 = wmma_bf16(pa0, frag_b(vr0),           o0);
    o1 = wmma_bf16(pa0, frag_b(vr0 + 16 * S_), o1);
    o2 = wmma_bf16(pa0, frag_b(vr0 + 32 * S_), o2);
    o3 = wmma_bf16(pa0, frag_b(vr0 + 48 * S_), o3);
    o0 = wmma_bf16(pa1, frag_b(vr1),           o0);
    o1 = wmma_bf16(pa1, frag_b(vr1 + 16 * S_), o1);
    o2 = wmma_bf16(pa1, frag_b(vr1 + 32 * S_), o2);
    o3 = wmma_bf16(pa1, frag_b(vr1 + 48 * S_), o3);
    __syncthreads();
  }
  if (lane < 16) slinv[wave][lane] = 1.0f / l_r;
  __syncthreads();
#pragma unroll
  for (int i = 0; i < 8; i++) {
    float inv = slinv[wave][i + half * 8];
    int s = q0 + i + half * 8;
    size_t rowoff = ((size_t)b * S_ + s) * E_ + h * D_ + l16;
    ao[rowoff +  0] = f2bf(o0[i] * inv);
    ao[rowoff + 16] = f2bf(o1[i] * inv);
    ao[rowoff + 32] = f2bf(o2[i] * inv);
    ao[rowoff + 48] = f2bf(o3[i] * inv);
  }
}

// ---- pass 3: output GEMM [8192x1024] x [1024x1024] + b_out -> fp32 ----------
__global__ void __launch_bounds__(256)
out_gemm_kernel(const bf16* __restrict__ a, const bf16* __restrict__ wt,
                const float* __restrict__ bias, float* __restrict__ out) {
  const int NT = E_ / 64;                        // 16 col blocks
  int lane = threadIdx.x & 31, wave = threadIdx.x >> 5;
  int half = lane >> 4, l16 = lane & 15;
  int tile = blockIdx.x * 8 + wave;
  int mT = tile / NT, nT = tile % NT;
  int m0 = mT * 32, n0 = nT * 64;
  const bf16* arow0 = a  + (size_t)(m0 + l16) * E_;
  const bf16* arow1 = a  + (size_t)(m0 + 16 + l16) * E_;
  const bf16* bptr  = wt + (size_t)(n0 + l16) * E_;
  v8f acc[2][4] = {};
#pragma unroll 2
  for (int k0 = 0; k0 < E_; k0 += 32) {
    v16bf a0 = frag_a(arow0 + k0, half);
    v16bf a1 = frag_a(arow1 + k0, half);
    const bf16* bp = bptr + k0 + half * 16;
    v16bf b0 = frag_b(bp);
    v16bf b1 = frag_b(bp + 16 * E_);
    v16bf b2 = frag_b(bp + 32 * E_);
    v16bf b3 = frag_b(bp + 48 * E_);
    acc[0][0] = wmma_bf16(a0, b0, acc[0][0]);
    acc[0][1] = wmma_bf16(a0, b1, acc[0][1]);
    acc[0][2] = wmma_bf16(a0, b2, acc[0][2]);
    acc[0][3] = wmma_bf16(a0, b3, acc[0][3]);
    acc[1][0] = wmma_bf16(a1, b0, acc[1][0]);
    acc[1][1] = wmma_bf16(a1, b1, acc[1][1]);
    acc[1][2] = wmma_bf16(a1, b2, acc[1][2]);
    acc[1][3] = wmma_bf16(a1, b3, acc[1][3]);
  }
#pragma unroll
  for (int mi = 0; mi < 2; mi++) {
#pragma unroll
    for (int j = 0; j < 4; j++) {
      int col = n0 + j * 16 + l16;
      float bv = bias[col];
#pragma unroll
      for (int i = 0; i < 8; i++) {
        int row = m0 + mi * 16 + i + half * 8;
        out[(size_t)row * E_ + col] = acc[mi][j][i] + bv;
      }
    }
  }
}

extern "C" void kernel_launch(void* const* d_in, const int* in_sizes, int n_in,
                              void* d_out, int out_size, void* d_ws,
                              size_t ws_size, hipStream_t stream) {
  (void)in_sizes; (void)n_in; (void)out_size; (void)ws_size;
  const float* x     = (const float*)d_in[0];
  const float* w_in  = (const float*)d_in[1];
  const float* b_in  = (const float*)d_in[2];
  const float* w_out = (const float*)d_in[3];
  const float* b_out = (const float*)d_in[4];
  float* out = (float*)d_out;

  char* ws = (char*)d_ws;
  size_t off = 0;
  bf16* xb    = (bf16*)(ws + off); off += (size_t)B_ * S_ * E_ * 2;   // 16MB
  bf16* wInT  = (bf16*)(ws + off); off += (size_t)E_ * N3_ * 2;       //  6MB
  bf16* wOutT = (bf16*)(ws + off); off += (size_t)E_ * E_ * 2;        //  2MB
  bf16* qq    = (bf16*)(ws + off); off += (size_t)B_ * S_ * E_ * 2;   // 16MB
  bf16* kk    = (bf16*)(ws + off); off += (size_t)B_ * S_ * E_ * 2;   // 16MB
  bf16* vT    = (bf16*)(ws + off); off += (size_t)B_ * S_ * E_ * 2;   // 16MB
  bf16* ao    = (bf16*)(ws + off); off += (size_t)B_ * S_ * E_ * 2;   // 16MB

  int nx4 = B_ * S_ * E_ / 4;
  cvt4_kernel<<<(nx4 + 255) / 256, 256, 0, stream>>>(x, xb, nx4);
  tcvt_kernel<<<(E_ * N3_ + 255) / 256, 256, 0, stream>>>(w_in, wInT, E_, N3_);
  tcvt_kernel<<<(E_ * E_ + 255) / 256, 256, 0, stream>>>(w_out, wOutT, E_, E_);

  qkv_gemm_kernel<<<(B_ * S_ / 32) * (N3_ / 64) / 8, 256, 0, stream>>>(
      xb, wInT, b_in, qq, kk, vT);
  attn_kernel<<<B_ * H_ * (S_ / 128), 256, 0, stream>>>(qq, kk, vT, ao);
  out_gemm_kernel<<<(B_ * S_ / 32) * (E_ / 64) / 8, 256, 0, stream>>>(
      ao, wOutT, b_out, out);
}
